// RNN_circular_2D_xy_9552007267010
// MI455X (gfx1250) — compile-verified, matrix operand order checked by reading, not run on hardware
//
#include <hip/hip_runtime.h>

// MI455X (gfx1250) persistent-workgroup RNN scan.
// B=256, T=1024, I=2, H=512. Batch-parallel: 16 workgroups x 16 batch rows.
// Per step each WG does a [16x512] @ [512x512]^T GEMM with v_wmma_f32_16x16x32_bf16,
// fp32 accumulate, fused relu + input-projection epilogue.
//
// Weight handling per wave (2 N-tiles, 16 K-tiles each):
//   - N-tile 0: 16 B-fragments pinned in VGPRs (128 regs), loaded once.
//   - N-tile 1: B-fragments streamed from L2 via inline-asm global_load_b128,
//     4-buffer / distance-2 software pipeline, explicit s_wait_loadcnt.
//   - A-fragments (h state in LDS) pipelined one k-tile ahead.
//   - __builtin_amdgcn_sched_barrier(0) fences each iteration's load phase from
//     its compute phase so the scheduler/RA cannot collapse the pipeline
//     (which previously re-created the 4x v_nop WMMA WAR bubbles and dscnt-0 stalls).

typedef __attribute__((ext_vector_type(16))) __bf16   v16bf;
typedef __attribute__((ext_vector_type(8)))  __bf16   v8bf;
typedef __attribute__((ext_vector_type(8)))  float    v8f;
typedef __attribute__((ext_vector_type(4)))  unsigned v4u;
typedef __attribute__((ext_vector_type(8)))  unsigned v8u;

#define BSZ      256
#define TLEN     1024
#define HDIM     512
#define NT_TILES 32        // H / 16  (N tiles)
#define KT_TILES 16        // H / 32  (K tiles per WMMA chain)
#define ROWPAD   520       // 512 + 8 bf16 pad: stride = 260 dwords = 4 (mod 64) -> conflict-free b128
#define MB       16        // batch rows per workgroup (one WMMA M tile)
#define NWAVES   16        // 512 threads / wave32

__device__ __forceinline__ unsigned short f32_to_bf16(float f) {
  unsigned u = __float_as_uint(f);
  unsigned r = u + 0x7FFFu + ((u >> 16) & 1u);   // round-to-nearest-even
  return (unsigned short)(r >> 16);
}

// ---------------------------------------------------------------------------
// Pre-pass: pack Wh^T (bf16) into the exact WMMA B-fragment register layout so
// the hot loop loads 32 contiguous bytes per lane per fragment.
// B fragment for 16x16x32 bf16 (K x N = 32 x 16), wave32:
//   lane L: n = L & 15, khalf = L >> 4
//   elem e<8  -> k = khalf*8 + e ; elem e>=8 -> k = 16 + khalf*8 + (e-8)
// B[k][n] = Wh[n][k]  (because the GEMM is h @ Wh^T).
// ---------------------------------------------------------------------------
__global__ void pack_wh_kernel(const float* __restrict__ Wh,
                               unsigned short* __restrict__ bp) {
  int idx = blockIdx.x * blockDim.x + threadIdx.x;
  if (idx >= NT_TILES * KT_TILES * 32) return;
  int lane  = idx & 31;
  int kt    = (idx >> 5) & (KT_TILES - 1);
  int nt    = idx >> 9;
  int n     = nt * 16 + (lane & 15);
  int khalf = lane >> 4;
  unsigned short* outp = bp + (size_t)idx * 16;
#pragma unroll
  for (int e = 0; e < 16; ++e) {
    int kin = (e < 8) ? (khalf * 8 + e) : (16 + khalf * 8 + (e - 8));
    int k   = kt * 32 + kin;
    outp[e] = f32_to_bf16(Wh[(size_t)n * HDIM + k]);
  }
}

// ---------------------------------------------------------------------------
// Persistent scan kernel: 16 blocks x 512 threads (16 waves).
// ---------------------------------------------------------------------------
__global__ __launch_bounds__(512, 1)
void rnn_scan_kernel(const float* __restrict__ x,              // [B, T, 2]
                     const unsigned short* __restrict__ bp,    // packed Wh bf16 fragments
                     const float* __restrict__ Wx,             // [H, 2]
                     const float* __restrict__ bx,             // [H]
                     float* __restrict__ out)                  // [T+1, B, H]
{
  __shared__ __align__(16) unsigned short hbuf[2][MB * ROWPAD]; // ping-pong h state (bf16)
  __shared__ float xbuf[MB * 2];                                // x[b, t, 0..1] for this tile

  const int tid   = threadIdx.x;
  const int lane  = tid & 31;
  const int wave  = tid >> 5;
  const int b0    = blockIdx.x * MB;

  const int ncol  = lane & 15;
  const int khalf = lane >> 4;
  const int nt0   = wave * 2;
  const int nt1   = wave * 2 + 1;
  const int n0    = nt0 * 16 + ncol;
  const int n1    = nt1 * 16 + ncol;

  // Per-lane constants for the fused input-projection epilogue.
  const float wx00 = Wx[n0 * 2 + 0], wx01 = Wx[n0 * 2 + 1], bx0v = bx[n0];
  const float wx10 = Wx[n1 * 2 + 0], wx11 = Wx[n1 * 2 + 1], bx1v = bx[n1];

  // h0: zeros except columns 0 and H/2 = 1. Write out[0] and seed hbuf[0].
  for (int i = tid; i < MB * HDIM; i += blockDim.x) {
    int m = i >> 9;           // / 512
    int n = i & (HDIM - 1);
    bool one = (n == 0) || (n == HDIM / 2);
    hbuf[0][m * ROWPAD + n] = one ? (unsigned short)0x3F80 : (unsigned short)0;
    out[(size_t)(b0 + m) * HDIM + n] = one ? 1.0f : 0.0f;
  }

  // Register-resident B fragments for n-tile 0 (16 frags x 8 VGPRs = 128 VGPRs).
  const unsigned short* bpw0 = bp + ((size_t)nt0 * KT_TILES) * 32 * 16;
  v16bf breg[KT_TILES];
#pragma unroll
  for (int kt = 0; kt < KT_TILES; ++kt)
    breg[kt] = *(const v16bf*)(bpw0 + ((size_t)kt * 32 + lane) * 16);

  // Streamed-fragment base address for n-tile 1 (per-lane 32B chunk).
  unsigned long long baddr =
      (unsigned long long)(bp + ((size_t)nt1 * KT_TILES) * 32 * 16) +
      (unsigned long long)lane * 32ull;

  __syncthreads();

  int p = 0;
#pragma unroll 1
  for (int t = 0; t < TLEN; ++t) {
    // Stage x[b, t, :] for this batch tile (32 floats) + prefetch next step.
    if (tid < MB * 2) {
      const float* xp = &x[((size_t)(b0 + (tid >> 1)) * TLEN + t) * 2 + (tid & 1)];
      xbuf[tid] = *xp;
      if (t + 1 < TLEN) __builtin_prefetch(xp + 2, 0, 1);  // global_prefetch_b8
    }
    __syncthreads();

    // A fragment source: lane L reads h row M = L & 15 (16-bit A layout,
    // K halves selected by lane>>4).
    const unsigned short* hrow = &hbuf[p][(lane & 15) * ROWPAD];

    v8f acc0 = {};
    v8f acc1 = {};

    // ---- streamed-B pipeline: 4 buffers, prefetch distance 2 ----
    v4u bb[4][2];
    {
      asm volatile("global_load_b128 %0, %1, off" : "=&v"(bb[0][0]) : "v"(baddr));
      asm volatile("global_load_b128 %0, %1, off" : "=&v"(bb[0][1]) : "v"(baddr + 16ull));
      unsigned long long a1 = baddr + 1024ull;
      asm volatile("global_load_b128 %0, %1, off" : "=&v"(bb[1][0]) : "v"(a1));
      asm volatile("global_load_b128 %0, %1, off" : "=&v"(bb[1][1]) : "v"(a1 + 16ull));
    }

    // ---- A pipeline: 3 rotating buffers, 1 k-tile ahead ----
    v8bf ab[3][2];
    ab[0][0] = *(const v8bf*)(hrow + khalf * 8);
    ab[0][1] = *(const v8bf*)(hrow + khalf * 8 + 16);

#pragma unroll
    for (int kt = 0; kt < KT_TILES; ++kt) {
      // ---------------- load phase (next tiles) ----------------
      if (kt + 2 < KT_TILES) {
        unsigned long long an = baddr + (unsigned long long)(kt + 2) * 1024ull;
        asm volatile("global_load_b128 %0, %1, off" : "=&v"(bb[(kt + 2) & 3][0]) : "v"(an));
        asm volatile("global_load_b128 %0, %1, off" : "=&v"(bb[(kt + 2) & 3][1]) : "v"(an + 16ull));
      }
      if (kt + 1 < KT_TILES) {
        const int cn = (kt + 1) * 32 + khalf * 8;
        ab[(kt + 1) % 3][0] = *(const v8bf*)(hrow + cn);        // ds_load_b128
        ab[(kt + 1) % 3][1] = *(const v8bf*)(hrow + cn + 16);   // ds_load_b128
      }
      // Fence: nothing may cross between load-issue and compute. Keeps the
      // next-tile buffers live across the WMMAs -> distinct registers -> no
      // WMMA WAR v_nops, and the dscnt wait covers a full-iteration-old load.
      __builtin_amdgcn_sched_barrier(0);

      // In-order completion: allow the younger (kt+1, kt+2) loads to stay in
      // flight; the k-tile kt pair is then guaranteed resident.
      if (kt + 2 < KT_TILES)       asm volatile("s_wait_loadcnt 0x4");
      else if (kt + 1 < KT_TILES)  asm volatile("s_wait_loadcnt 0x2");
      else                         asm volatile("s_wait_loadcnt 0x0");

      v8u cu = __builtin_shufflevector(bb[kt & 3][0], bb[kt & 3][1],
                                       0, 1, 2, 3, 4, 5, 6, 7);
      v16bf bf1 = __builtin_bit_cast(v16bf, cu);
      v16bf a   = __builtin_shufflevector(ab[kt % 3][0], ab[kt % 3][1],
                                          0,1,2,3,4,5,6,7,8,9,10,11,12,13,14,15);

      acc0 = __builtin_amdgcn_wmma_f32_16x16x32_bf16(false, a, false, breg[kt],
                                                     (short)0, acc0, false, false);
      acc1 = __builtin_amdgcn_wmma_f32_16x16x32_bf16(false, a, false, bf1,
                                                     (short)0, acc1, false, false);
      __builtin_amdgcn_sched_barrier(0);
    }

    // Epilogue: C/D layout is lane L, VGPR r -> M = r + 8*(L>>4), N = L&15.
    unsigned short* hn = hbuf[p ^ 1];
    float* orow = out + (size_t)(t + 1) * BSZ * HDIM + (size_t)b0 * HDIM;
#pragma unroll
    for (int r = 0; r < 8; ++r) {
      int m = r + 8 * (lane >> 4);
      float x0 = xbuf[m * 2 + 0];
      float x1 = xbuf[m * 2 + 1];
      float v0 = acc0[r] + x0 * wx00 + x1 * wx01 + bx0v;
      float v1 = acc1[r] + x0 * wx10 + x1 * wx11 + bx1v;
      v0 = v0 > 0.0f ? v0 : 0.0f;
      v1 = v1 > 0.0f ? v1 : 0.0f;
      __builtin_nontemporal_store(v0, &orow[(size_t)m * HDIM + n0]);  // 537MB write-once stream
      __builtin_nontemporal_store(v1, &orow[(size_t)m * HDIM + n1]);
      hn[m * ROWPAD + n0] = f32_to_bf16(v0);
      hn[m * ROWPAD + n1] = f32_to_bf16(v1);
    }
    __syncthreads();
    p ^= 1;
  }
}

extern "C" void kernel_launch(void* const* d_in, const int* in_sizes, int n_in,
                              void* d_out, int out_size, void* d_ws, size_t ws_size,
                              hipStream_t stream) {
  const float* x  = (const float*)d_in[0];   // [256, 1024, 2]
  const float* Wh = (const float*)d_in[1];   // [512, 512]
  const float* Wx = (const float*)d_in[2];   // [512, 2]
  const float* bx = (const float*)d_in[3];   // [512]
  float* out = (float*)d_out;                // [1025, 256, 512]

  unsigned short* bp = (unsigned short*)d_ws;  // 512 KB packed Wh bf16 fragments

  {
    int total = NT_TILES * KT_TILES * 32;      // 16384 pack threads
    pack_wh_kernel<<<(total + 255) / 256, 256, 0, stream>>>(Wh, bp);
  }
  rnn_scan_kernel<<<BSZ / MB, NWAVES * 32, 0, stream>>>(x, bp, Wx, bx, out);
}
